// SioConvPSLayer_13709535609451
// MI455X (gfx1250) — matrix operand order repaired and checked by reading, not compiled, not run on hardware
//
#include <hip/hip_runtime.h>
#include <cstdint>
#include <cstddef>

// ---------------------------------------------------------------------------
// SioConvPS layer for MI455X (gfx1250, wave32, WMMA).
//   B=4, L=8192, D=1024.
//   GEMMs (275 GFLOP total) run on v_wmma_f32_16x16x32_bf16; each wave owns a
//   32x64 output strip (8 WMMAs per K-step vs 12 b128 loads).
//   Sequential scan over L is parallelized as a 3-phase chunked scan.
// ---------------------------------------------------------------------------

typedef __bf16 bf16;
typedef __attribute__((ext_vector_type(16))) __bf16 v16bf;
typedef __attribute__((ext_vector_type(8)))  __bf16 v8bf;
typedef __attribute__((ext_vector_type(8)))  float  v8f;

#define BDIM 4
#define LDIM 8192
#define DDIM 1024
#define BLROWS (BDIM * LDIM)   /* 32768 GEMM rows            */
#define NCHUNK 64              /* scan chunks along L        */
#define CHLEN  (LDIM / NCHUNK) /* 128 elements per chunk     */

// log_sigmoid(x) = min(x,0) - log1p(exp(-|x|))
__device__ __forceinline__ float logsigf(float x) {
  return fminf(x, 0.0f) - log1pf(expf(-fabsf(x)));
}

// ---------------------------------------------------------------------------
// fp32 -> bf16 conversion (grid-stride)
// ---------------------------------------------------------------------------
__global__ void cvt_f32_bf16(const float* __restrict__ in,
                             bf16* __restrict__ out, int n) {
  for (int i = blockIdx.x * blockDim.x + threadIdx.x; i < n;
       i += gridDim.x * blockDim.x)
    out[i] = (bf16)in[i];
}

// ---------------------------------------------------------------------------
// WMMA GEMM:  out[M,N] = A[M,K] @ W[N,K]^T + bias     (torch Linear layout)
// Each wave computes a 32(M) x 64(N) strip: 2 A fragments x 4 B fragments,
// 8 accumulators, B fragments reused across both M-tiles.
// Fragment layouts used here (contiguous k order, same for A and B):
//   A 16x32 bf16 : lane (m=lane&15) holds K = (lane>>4)*16 + {0..15}
//   B 32x16 bf16 : lane (n=lane&15) holds K = (lane>>4)*16 + {0..15}
//   C/D 16x16    : vgpr i, lane -> (m = i + 8*(lane>=16), n = lane&15)
// EPI: 0 = bias only; 1 = bias then * silu(gate); 2 = log_sigmoid(bias-added)
// ---------------------------------------------------------------------------
template <int EPI>
__global__ __launch_bounds__(256) void gemm_wmma(
    const bf16* __restrict__ A, const bf16* __restrict__ W,
    const float* __restrict__ bias, const float* __restrict__ gate,
    float* __restrict__ out, int M, int N, int K) {
  const int lane = threadIdx.x & 31;
  const int wave = threadIdx.x >> 5;
  const int gw = blockIdx.x * 8 + wave;
  const int nStrips = N >> 6;               // strips of 64 columns
  const int m0 = (gw / nStrips) * 32;
  const int n0 = (gw % nStrips) * 64;
  if (m0 >= M) return;                      // wave-uniform (EXEC stays all-1)

  const int rc  = lane & 15;
  const int kbA = (lane >> 4) * 16;         // A k-offset per lane half
  const int kbB = (lane >> 4) * 16;         // B k-offset per lane half
  const bf16* ap0 = A + (size_t)(m0 + rc) * K + kbA;
  const bf16* ap1 = A + (size_t)(m0 + 16 + rc) * K + kbA;
  const bf16* wp  = W + (size_t)(n0 + rc) * K + kbB;

  v8f acc[2][4];
#pragma unroll
  for (int r = 0; r < 2; ++r)
#pragma unroll
    for (int t = 0; t < 4; ++t) acc[r][t] = (v8f){0, 0, 0, 0, 0, 0, 0, 0};

  for (int k0 = 0; k0 < K; k0 += 32) {
    v8bf a0lo = *(const v8bf*)(ap0 + k0);
    v8bf a0hi = *(const v8bf*)(ap0 + k0 + 8);
    v8bf a1lo = *(const v8bf*)(ap1 + k0);
    v8bf a1hi = *(const v8bf*)(ap1 + k0 + 8);
    v16bf a0 = __builtin_shufflevector(a0lo, a0hi, 0, 1, 2, 3, 4, 5, 6, 7, 8,
                                       9, 10, 11, 12, 13, 14, 15);
    v16bf a1 = __builtin_shufflevector(a1lo, a1hi, 0, 1, 2, 3, 4, 5, 6, 7, 8,
                                       9, 10, 11, 12, 13, 14, 15);
    __builtin_prefetch(ap0 + k0 + 32, 0, 3);   // WGP-scope prefetch
    __builtin_prefetch(ap1 + k0 + 32, 0, 3);
#pragma unroll
    for (int t = 0; t < 4; ++t) {
      v16bf b = *(const v16bf*)(wp + (size_t)(t * 16) * K + k0);
      acc[0][t] = __builtin_amdgcn_wmma_f32_16x16x32_bf16(
          false, a0, false, b, (short)0, acc[0][t], false, false);
      acc[1][t] = __builtin_amdgcn_wmma_f32_16x16x32_bf16(
          false, a1, false, b, (short)0, acc[1][t], false, false);
    }
  }

#pragma unroll
  for (int r = 0; r < 2; ++r) {
    const int rbase = m0 + r * 16 + ((lane >> 4) << 3);  // +8 upper lane half
#pragma unroll
    for (int t = 0; t < 4; ++t) {
      const int col = n0 + t * 16 + rc;
      const float bv = bias[col];
#pragma unroll
      for (int i = 0; i < 8; ++i) {
        const int row = rbase + i;
        float v = acc[r][t][i] + bv;
        if constexpr (EPI == 1) {
          float g = gate[(size_t)row * N + col];
          v *= g / (1.0f + expf(-g));            // silu gate
        } else if constexpr (EPI == 2) {
          v = logsigf(v);                        // fused log_sigmoid
        }
        out[(size_t)row * N + col] = v;
      }
    }
  }
}

// ---------------------------------------------------------------------------
// Scan phase 1: per-(b,chunk,d) local partials.
//   c_part = sum of log_sigmoid(-dt) over chunk
//   l_part = LSE over chunk of (ln_z + ln_da - local_cumsum)
// ln_z arrives pre-activated (EPI==2 in its GEMM).  ln_da = dt + ln_o_da.
// ---------------------------------------------------------------------------
__global__ void scan_partials(const float* __restrict__ dtp,
                              const float* __restrict__ lnz,
                              float* __restrict__ c_part,
                              float* __restrict__ l_part) {
  const int tid = blockIdx.x * blockDim.x + threadIdx.x;
  if (tid >= BDIM * NCHUNK * DDIM) return;
  const int d = tid & (DDIM - 1);
  const int ch = (tid >> 10) & (NCHUNK - 1);
  const int b = tid >> 16;  // NCHUNK*DDIM = 65536
  const size_t base = ((size_t)b * LDIM + (size_t)ch * CHLEN) * DDIM + d;
  float c = 0.0f, m = -INFINITY, s = 0.0f;
  for (int l = 0; l < CHLEN; ++l) {
    const size_t idx = base + (size_t)l * DDIM;
    const float dt = dtp[idx];
    const float lod = logsigf(-dt);
    const float lda = dt + lod;      // log_sigmoid(dt) = dt + log_sigmoid(-dt)
    c += lod;
    const float v = lnz[idx] + lda - c;
    const float mn = fmaxf(m, v);
    s = s * expf(m - mn) + expf(v - mn);
    m = mn;
  }
  c_part[tid] = c;
  l_part[tid] = (s > 0.0f) ? (m + logf(s)) : -INFINITY;
}

// ---------------------------------------------------------------------------
// Scan phase 2: exclusive prefix over the 64 chunks per (b,d) channel.
// Combine: (c, L) o (c', L') = (c + c', logaddexp(L, L' - c)).
// ---------------------------------------------------------------------------
__global__ void scan_prefix(const float* __restrict__ c_part,
                            const float* __restrict__ l_part,
                            float* __restrict__ c_pre,
                            float* __restrict__ l_pre) {
  const int tid = blockIdx.x * blockDim.x + threadIdx.x;  // B*D = 4096
  if (tid >= BDIM * DDIM) return;
  const int d = tid & (DDIM - 1);
  const int b = tid >> 10;
  float rc = 0.0f, rl = -INFINITY;
  for (int ch = 0; ch < NCHUNK; ++ch) {
    const int idx = (b * NCHUNK + ch) * DDIM + d;
    c_pre[idx] = rc;
    l_pre[idx] = rl;
    const float lp = l_part[idx] - rc;         // rebase chunk LSE to global
    const float mn = fmaxf(rl, lp);
    rl = (mn == -INFINITY) ? -INFINITY
                           : mn + log1pf(expf(fminf(rl, lp) - mn));
    rc += c_part[idx];
  }
}

// ---------------------------------------------------------------------------
// Scan phase 3: re-run local scan seeded with global prefixes, emit h (bf16).
//   h = exp(lcse + cumsum) + exp(cumsum) * h0[d]
// ---------------------------------------------------------------------------
__global__ void scan_apply(const float* __restrict__ dtp,
                           const float* __restrict__ lnz,
                           const float* __restrict__ c_pre,
                           const float* __restrict__ l_pre,
                           const float* __restrict__ h0,
                           bf16* __restrict__ hout) {
  const int tid = blockIdx.x * blockDim.x + threadIdx.x;
  if (tid >= BDIM * NCHUNK * DDIM) return;
  const int d = tid & (DDIM - 1);
  const int ch = (tid >> 10) & (NCHUNK - 1);
  const int b = tid >> 16;
  const size_t base = ((size_t)b * LDIM + (size_t)ch * CHLEN) * DDIM + d;
  float c = c_pre[tid];
  float m = l_pre[tid];
  float s = (m == -INFINITY) ? 0.0f : 1.0f;
  const float h0v = h0[d];
  for (int l = 0; l < CHLEN; ++l) {
    const size_t idx = base + (size_t)l * DDIM;
    const float dt = dtp[idx];
    const float lod = logsigf(-dt);
    const float lda = dt + lod;
    c += lod;
    const float v = lnz[idx] + lda - c;
    const float mn = fmaxf(m, v);
    s = s * expf(m - mn) + expf(v - mn);
    m = mn;
    const float lcse = m + logf(s);
    const float h = expf(lcse + c) + expf(c) * h0v;
    hout[idx] = (bf16)h;
  }
}

// ---------------------------------------------------------------------------
// Host-side orchestration
// ---------------------------------------------------------------------------
extern "C" void kernel_launch(void* const* d_in, const int* in_sizes, int n_in,
                              void* d_out, int out_size, void* d_ws,
                              size_t ws_size, hipStream_t stream) {
  const float* x     = (const float*)d_in[0];
  const float* w_lnz = (const float*)d_in[1];
  const float* b_lnz = (const float*)d_in[2];
  const float* w_dt  = (const float*)d_in[3];
  const float* b_dt  = (const float*)d_in[4];
  const float* w_y   = (const float*)d_in[5];
  const float* b_y   = (const float*)d_in[6];
  const float* w_ya  = (const float*)d_in[7];
  const float* b_ya  = (const float*)d_in[8];
  const float* h0    = (const float*)d_in[9];
  float* out = (float*)d_out;

  const size_t nBLD = (size_t)BLROWS * DDIM;  // 33,554,432
  const size_t nDD  = (size_t)DDIM * DDIM;    //  1,048,576
  const size_t nPart = (size_t)BDIM * NCHUNK * DDIM;  // 262,144

  // Workspace layout (bytes):
  char* ws = (char*)d_ws;
  size_t o = 0;
  bf16* x_bf     = (bf16*)(ws + o); o += nBLD * sizeof(bf16);   //  64 MiB
  bf16* wlnz_bf  = (bf16*)(ws + o); o += nDD * sizeof(bf16);    //   2 MiB
  bf16* wdt_bf   = (bf16*)(ws + o); o += nDD * sizeof(bf16);
  bf16* wya_bf   = (bf16*)(ws + o); o += nDD * sizeof(bf16);
  bf16* wy_bf    = (bf16*)(ws + o); o += nDD * sizeof(bf16);
  float* dt_pre  = (float*)(ws + o); o += nBLD * sizeof(float); // 128 MiB
  float* lnz_act = (float*)(ws + o); o += nBLD * sizeof(float); // 128 MiB
  float* ya_pre  = (float*)(ws + o); o += nBLD * sizeof(float); // 128 MiB
  bf16* h_bf     = (bf16*)(ws + o); o += nBLD * sizeof(bf16);   //  64 MiB
  float* c_part  = (float*)(ws + o); o += nPart * sizeof(float);
  float* l_part  = (float*)(ws + o); o += nPart * sizeof(float);
  float* c_pre   = (float*)(ws + o); o += nPart * sizeof(float);
  float* l_pre   = (float*)(ws + o); o += nPart * sizeof(float);
  (void)ws_size; (void)in_sizes; (void)n_in; (void)out_size;

  // 1) downconvert activations + weights to bf16 for the matrix path
  cvt_f32_bf16<<<8192, 256, 0, stream>>>(x, x_bf, (int)nBLD);
  cvt_f32_bf16<<<2048, 256, 0, stream>>>(w_lnz, wlnz_bf, (int)nDD);
  cvt_f32_bf16<<<2048, 256, 0, stream>>>(w_dt,  wdt_bf,  (int)nDD);
  cvt_f32_bf16<<<2048, 256, 0, stream>>>(w_ya,  wya_bf,  (int)nDD);
  cvt_f32_bf16<<<2048, 256, 0, stream>>>(w_y,   wy_bf,   (int)nDD);

  // 2) three input-projection GEMMs (share x_bf): 32x64 strip per wave.
  //    waves = (M/32)*(N/64) = 1024*16 = 16384 -> 2048 blocks of 8 waves.
  const int gemmBlocks = (BLROWS / 32) * (DDIM / 64) / 8;
  gemm_wmma<0><<<gemmBlocks, 256, 0, stream>>>(x_bf, wdt_bf, b_dt, nullptr,
                                               dt_pre, BLROWS, DDIM, DDIM);
  gemm_wmma<2><<<gemmBlocks, 256, 0, stream>>>(x_bf, wlnz_bf, b_lnz, nullptr,
                                               lnz_act, BLROWS, DDIM, DDIM);
  gemm_wmma<0><<<gemmBlocks, 256, 0, stream>>>(x_bf, wya_bf, b_ya, nullptr,
                                               ya_pre, BLROWS, DDIM, DDIM);

  // 3) chunked scan over L (logcumsumexp + cumsum), emits h in bf16
  scan_partials<<<(int)(nPart / 256), 256, 0, stream>>>(dt_pre, lnz_act,
                                                        c_part, l_part);
  scan_prefix<<<(BDIM * DDIM) / 256, 256, 0, stream>>>(c_part, l_part,
                                                       c_pre, l_pre);
  scan_apply<<<(int)(nPart / 256), 256, 0, stream>>>(dt_pre, lnz_act, c_pre,
                                                     l_pre, h0, h_bf);

  // 4) output projection with fused silu gate: y = (h@W_y^T+b_y)*silu(ya_pre)
  gemm_wmma<1><<<gemmBlocks, 256, 0, stream>>>(h_bf, wy_bf, b_y, ya_pre, out,
                                               BLROWS, DDIM, DDIM);
}
